// AddrNet_29454885715970
// MI455X (gfx1250) — matrix operand order repaired and checked by reading, not compiled, not run on hardware
//
#include <hip/hip_runtime.h>
#include <hip/hip_bf16.h>
#include <math.h>

// ---------------------------------------------------------------------------
// AddrNet forward (eval argmax path) for MI455X / gfx1250, wave32 + WMMA bf16.
// One wave = one 16-row batch tile. W_out resident in WMMA B-layout registers
// (128 VGPRs, reused 8x16 WMMAs per tile via grid-stride persistence).
// Biases fused into the zero-padded K=16 row of the B operands with a 1.0 in
// the A operand at K=16, so every logits/mlp WMMA uses inline C=0.
// W_in / W_out operands prebuilt once per block in LDS (conflict-free b64).
// h carried in fp32 lane registers; LDS for per-wave 16x16 transposes and the
// 16-way argmax reduce (same-wave in-order DS + s_wait_dscnt, no block sync).
// ---------------------------------------------------------------------------

#define B_ROWS  524288
#define D_MODEL 128
#define HID     16
#define N_BINS  256
#define DEPTH   8
#define NT      (N_BINS / 16)     // 16 N-tiles of the logits GEMM
#define WAVES   8                 // 256-thread block = 8 wave32
#define TILES   (B_ROWS / 16)     // 32768 row tiles

typedef __attribute__((ext_vector_type(16))) __bf16 v16bf;
typedef __attribute__((ext_vector_type(8)))  float  v8f;

union PkB64 { unsigned long long u; __bf16 h[4]; };

__device__ __forceinline__ v8f wmma_bf16(v16bf a, v16bf b, v8f c) {
  // D(f32 16x16) = A(bf16 16x32) * B(bf16 32x16) + C
  return __builtin_amdgcn_wmma_f32_16x16x32_bf16(
      /*neg_a=*/false, a, /*neg_b=*/false, b,
      /*c_mod=*/(short)0, c, /*reuse_a=*/false, /*reuse_b=*/false);
}

// LDS ops are in-order within a wave; hw wait + compiler barrier makes
// same-wave cross-lane LDS store->load sequences safe without a workgroup
// barrier.
__device__ __forceinline__ void wave_lds_fence() {
  asm volatile("s_wait_dscnt 0" ::: "memory");
}

__global__ __launch_bounds__(256) void addrnet_fwd(
    const float* __restrict__ hidden,   // [B, 128]
    const float* __restrict__ W_in,     // [128, 16]
    const float* __restrict__ b_in,     // [16]
    const float* __restrict__ embed,    // [256, 16]
    const float* __restrict__ W_mlp,    // [16, 16]
    const float* __restrict__ b_mlp,    // [16]
    const float* __restrict__ W_out,    // [16, 256]
    const float* __restrict__ b_out,    // [256]
    float*       __restrict__ out)      // [B, 8] bin indices (as float)
{
  // Per-wave LDS scratch (no cross-wave sharing).
  __shared__ float s_tr[WAVES][16][17];   // 16x16 f32 transpose buffer (pad 17)
  __shared__ float s_pv[WAVES][16][17];   // argmax partial values
  __shared__ int   s_pi[WAVES][16][17];   // argmax partial indices
  __shared__ int   s_ix[WAVES][16];       // final per-row argmax
  // Block-shared prebuilt WMMA B operands (entry = [blk][pair p][lane], b64,
  // lane stride 8B -> conflict-free ds_load_b64).
  __shared__ unsigned long long s_bin [4  * 4 * 32];  // W_in  (K=0..127)
  __shared__ unsigned long long s_wout[NT * 4 * 32];  // W_out (+bias @ K=16)

  const int tid  = threadIdx.x;
  const int w    = tid >> 5;
  const int lane = tid & 31;
  const int half = lane >> 4;   // 16-lane group
  const int n    = lane & 15;   // A row index / B,C column index

  // ---------------------------------------------------------------------
  // Cooperative one-time build of B operands (ISA 7.12.2 layout):
// B (32x16 bf16): element i<8 -> K=i+8*half, element i>=8 -> K=i+8+8*half
//                 (K>=16 zero pad; lanes 0-15 element 8 == K=16 carries
//                 the bias row for the fused-bias trick).
  // ---------------------------------------------------------------------
  for (int e = tid; e < 4 * 4 * 32; e += 256) {       // W_in: all K real
    const int kb = e >> 7, p = (e >> 5) & 3, l = e & 31;
    const int lh = l >> 4, ln = l & 15;
    PkB64 pk;
    #pragma unroll
    for (int q = 0; q < 4; ++q) {
      const int K = kb * 32 + (p >> 1) * 16 + lh * 8 + 4 * (p & 1) + q;
      pk.h[q] = (__bf16)W_in[K * HID + ln];
    }
    s_bin[e] = pk.u;
  }
  for (int e = tid; e < NT * 4 * 32; e += 256) {      // W_out + fused b_out
    const int t = e >> 7, p = (e >> 5) & 3, l = e & 31;
    const int lh = l >> 4, ln = l & 15;
    PkB64 pk;
    if (p < 2) {
      #pragma unroll
      for (int q = 0; q < 4; ++q)
        pk.h[q] = (__bf16)W_out[(8 * lh + 4 * p + q) * N_BINS + 16 * t + ln];
    } else {
      #pragma unroll
      for (int q = 0; q < 4; ++q) pk.h[q] = (__bf16)0.0f;
      if (lh == 0 && p == 2) pk.h[0] = (__bf16)b_out[16 * t + ln];   // K=16 bias row
    }
    s_wout[e] = pk.u;
  }
  __syncthreads();

  // ---------------------------------------------------------------------
  // Per-wave resident operands.
  // ---------------------------------------------------------------------
  v16bf Bout[NT];
  #pragma unroll
  for (int t = 0; t < NT; ++t) {
    v16bf b;
    #pragma unroll
    for (int p = 0; p < 4; ++p) {
      PkB64 pk; pk.u = s_wout[t * 128 + p * 32 + lane];
      b[4*p+0] = pk.h[0]; b[4*p+1] = pk.h[1];
      b[4*p+2] = pk.h[2]; b[4*p+3] = pk.h[3];
    }
    Bout[t] = b;
  }

  v16bf Bmlp;   // W_mlp + fused b_mlp @ K=16
  {
    v16bf b;
    #pragma unroll
    for (int i = 0; i < 16; ++i) b[i] = (__bf16)0.0f;
    #pragma unroll
    for (int i = 0; i < 8; ++i) b[i] = (__bf16)W_mlp[(i + 8 * half) * HID + n];
    if (half == 0) {
      b[8] = (__bf16)b_mlp[n];
    }
    Bmlp = b;
  }

  const float  bInR  = b_in[n];
  const __bf16 kOne  = (half == 0) ? (__bf16)1.0f : (__bf16)0.0f;  // A @ K=16
  const __bf16 kZero = (__bf16)0.0f;

  // ---------------------------------------------------------------------
  // Grid-stride over 16-row tiles (wave-uniform => EXEC all-1s at WMMAs).
  // ---------------------------------------------------------------------
  const int nWaves = (int)((gridDim.x * blockDim.x) >> 5);
  const int gw     = (int)((blockIdx.x * blockDim.x + tid) >> 5);

  for (int tile = gw; tile < TILES; tile += nWaves) {
    const int rowBase = tile * 16;

    // ------------------- input projection: h = X @ W_in + b_in ----------
    v8f c;
    #pragma unroll
    for (int j = 0; j < 8; ++j) c[j] = bInR;

    const float* xrow = hidden + (size_t)(rowBase + n) * D_MODEL;
    #pragma unroll
    for (int kb = 0; kb < 4; ++kb) {
      // A layout (bf16 16x32): elem i<8 -> K = i + 8*half,
      //                        elem i>=8 -> K = i + 8 + 8*half.
      const float* p0 = xrow + kb * 32 + 8 * half;
      const float* p1 = p0 + 16;
      float4 q0 = *(const float4*)(p0);
      float4 q1 = *(const float4*)(p0 + 4);
      float4 q2 = *(const float4*)(p1);
      float4 q3 = *(const float4*)(p1 + 4);
      v16bf a;
      a[0]=(__bf16)q0.x;  a[1]=(__bf16)q0.y;  a[2]=(__bf16)q0.z;  a[3]=(__bf16)q0.w;
      a[4]=(__bf16)q1.x;  a[5]=(__bf16)q1.y;  a[6]=(__bf16)q1.z;  a[7]=(__bf16)q1.w;
      a[8]=(__bf16)q2.x;  a[9]=(__bf16)q2.y;  a[10]=(__bf16)q2.z; a[11]=(__bf16)q2.w;
      a[12]=(__bf16)q3.x; a[13]=(__bf16)q3.y; a[14]=(__bf16)q3.z; a[15]=(__bf16)q3.w;

      v16bf bw;                                   // W_in operand from LDS
      #pragma unroll
      for (int p = 0; p < 4; ++p) {
        PkB64 pk; pk.u = s_bin[kb * 128 + p * 32 + lane];
        bw[4*p+0] = pk.h[0]; bw[4*p+1] = pk.h[1];
        bw[4*p+2] = pk.h[2]; bw[4*p+3] = pk.h[3];
      }
      c = wmma_bf16(a, bw, c);
    }

    // C layout -> per-lane "A-row" fp32 h: hA[i] = h[m=n][i + 8*half]
    float hA[8];
    #pragma unroll
    for (int j = 0; j < 8; ++j) s_tr[w][j + 8 * half][n] = c[j];
    wave_lds_fence();
    #pragma unroll
    for (int i = 0; i < 8; ++i) hA[i] = s_tr[w][n][i + 8 * half];
    wave_lds_fence();

    // ------------------------------- depth loop -------------------------
    for (int d = 0; d < DEPTH; ++d) {
      // Pack h into A operand; element 8 = 1.0 at K=16 activates the fused
      // bias row of the B operands; K=17..31 stays zero.
      v16bf ah;
      #pragma unroll
      for (int i = 0; i < 8; ++i) ah[i] = (__bf16)hA[i];
      ah[8] = kOne;
      #pragma unroll
      for (int i = 9; i < 16; ++i) ah[i] = kZero;

      // logits = [h,1] @ [W_out;b_out], fused per-lane running argmax.
      float bestV[8];
      int   bestI[8];
      #pragma unroll
      for (int j = 0; j < 8; ++j) { bestV[j] = -3.402823466e38f; bestI[j] = 0; }

      #pragma unroll
      for (int t = 0; t < NT; ++t) {
        v8f zc = {0.f, 0.f, 0.f, 0.f, 0.f, 0.f, 0.f, 0.f};
        v8f cc = wmma_bf16(ah, Bout[t], zc);      // C = inline 0
        const int idx = 16 * t + n;               // ascending t => min idx ties
        #pragma unroll
        for (int j = 0; j < 8; ++j) {
          if (cc[j] > bestV[j]) { bestV[j] = cc[j]; bestI[j] = idx; }
        }
      }

      // Cross-lane argmax over the 16 column classes (per-wave LDS).
      #pragma unroll
      for (int j = 0; j < 8; ++j) {
        s_pv[w][j + 8 * half][n] = bestV[j];
        s_pi[w][j + 8 * half][n] = bestI[j];
      }
      wave_lds_fence();
      if (lane < 16) {
        float bv = s_pv[w][lane][0];
        int   bi = s_pi[w][lane][0];
        #pragma unroll
        for (int cix = 1; cix < 16; ++cix) {
          float v  = s_pv[w][lane][cix];
          int   id = s_pi[w][lane][cix];
          if (v > bv || (v == bv && id < bi)) { bv = v; bi = id; }
        }
        s_ix[w][lane] = bi;
        out[(size_t)(rowBase + lane) * DEPTH + d] = (float)bi;
      }
      wave_lds_fence();

      // h += embed[idx]   (embed row = 64B; two coalesced float4 per lane)
      const int myIdx = s_ix[w][n];
      const float* e = embed + myIdx * HID + 8 * half;
      float4 e0 = *(const float4*)(e);
      float4 e1 = *(const float4*)(e + 4);
      hA[0] += e0.x; hA[1] += e0.y; hA[2] += e0.z; hA[3] += e0.w;
      hA[4] += e1.x; hA[5] += e1.y; hA[6] += e1.z; hA[7] += e1.w;

      // h = silu([h,1] @ [W_mlp;b_mlp])
      v16bf ah2;
      #pragma unroll
      for (int i = 0; i < 8; ++i) ah2[i] = (__bf16)hA[i];
      ah2[8] = kOne;
      #pragma unroll
      for (int i = 9; i < 16; ++i) ah2[i] = kZero;

      v8f zm = {0.f, 0.f, 0.f, 0.f, 0.f, 0.f, 0.f, 0.f};
      v8f cm = wmma_bf16(ah2, Bmlp, zm);          // C = inline 0

      // SiLU then transpose C layout back to per-lane A-row fp32 h.
      #pragma unroll
      for (int j = 0; j < 8; ++j) {
        float z = cm[j];
        float s = 1.0f / (1.0f + __expf(-z));
        s_tr[w][j + 8 * half][n] = z * s;
      }
      wave_lds_fence();
      #pragma unroll
      for (int i = 0; i < 8; ++i) hA[i] = s_tr[w][n][i + 8 * half];
      wave_lds_fence();
    }
  }
}

extern "C" void kernel_launch(void* const* d_in, const int* in_sizes, int n_in,
                              void* d_out, int out_size, void* d_ws, size_t ws_size,
                              hipStream_t stream) {
  (void)in_sizes; (void)n_in; (void)out_size; (void)d_ws; (void)ws_size;
  const float* hidden = (const float*)d_in[0];
  const float* W_in   = (const float*)d_in[1];
  const float* b_in   = (const float*)d_in[2];
  const float* embed  = (const float*)d_in[3];
  const float* W_mlp  = (const float*)d_in[4];
  const float* b_mlp  = (const float*)d_in[5];
  const float* W_out  = (const float*)d_in[6];
  const float* b_out  = (const float*)d_in[7];
  float* out = (float*)d_out;

  // 1024 blocks x 256 threads = 8192 waves; 32768 tiles => exactly 4/wave.
  addrnet_fwd<<<dim3(1024), dim3(256), 0, stream>>>(
      hidden, W_in, b_in, embed, W_mlp, b_mlp, W_out, b_out, out);
}
